// GraphConv_46660524704516
// MI455X (gfx1250) — matrix operand order repaired and checked by reading, not compiled, not run on hardware
//
#include <hip/hip_runtime.h>
#include <hip/hip_bf16.h>

// ---------------------------------------------------------------------------
// GraphSAGE layer for MI455X (gfx1250, wave32, WMMA)
//   six fused GEMM+BN-stat kernels (bf16 WMMA, f32 accum), BN-apply+ReLU fused
//   into the next GEMM's A-stage, concat+segment-mean fused into comb GEMM,
//   one-wave-per-edge scatter with hardware f32 atomics.
// ---------------------------------------------------------------------------

#define D128      128
#define TILE_ROWS 128
#define KCHUNK    64
#define LDSTR     72     // padded LDS K-stride (144B rows: 16B-aligned, spreads banks)

typedef __attribute__((ext_vector_type(16))) __bf16 v16bf;
typedef __attribute__((ext_vector_type(8)))  __bf16 v8bf;
typedef __attribute__((ext_vector_type(8)))  float  v8f;

__global__ void zero_kernel(float* __restrict__ p, long long n) {
  long long i = (long long)blockIdx.x * 256 + threadIdx.x;
  if (i < n) p[i] = 0.f;
}

// Fused GEMM: Y = op(A) @ W + bias. op = per-column BN+ReLU of previous layer
// (sc0/sh0; nullptr -> identity). If A1 != nullptr, columns [128,256) of the
// operand come from A1 (segment sums) / max(cnt,1) (fused concat+mean).
// Also accumulates per-column sum / sum-of-squares for this layer's BN.
__global__ __launch_bounds__(256)
void gemm_bn_kernel(const float* __restrict__ A0,
                    const float* __restrict__ sc0, const float* __restrict__ sh0,
                    const float* __restrict__ A1, const float* __restrict__ cntv,
                    const float* __restrict__ W,  const float* __restrict__ bias,
                    float* __restrict__ Yp,
                    float* __restrict__ gsum, float* __restrict__ gsumsq,
                    int N, int K)
{
  __shared__ __bf16 AL[TILE_ROWS * LDSTR];   // 18 KB  A tile (rows x Kchunk)
  __shared__ __bf16 WL[D128 * LDSTR];        // 18 KB  W^T tile (cols x Kchunk)
  __shared__ float  biasL[D128];
  __shared__ float  colS[D128];
  __shared__ float  colSS[D128];

  const int tid     = threadIdx.x;
  const int rowBase = blockIdx.x * TILE_ROWS;
  if (tid < D128) { biasL[tid] = bias[tid]; colS[tid] = 0.f; colSS[tid] = 0.f; }

  const int wave = tid >> 5;
  const int lane = tid & 31;
  const int l15  = lane & 15;
  const int hi   = lane >> 4;          // lane half selector

  v8f acc[8] = {};                     // 8 col-tiles of 16x16 f32

  for (int kc = 0; kc < K; kc += KCHUNK) {
    // ---- stage A chunk: float4 global loads, BN/mean fused, 8B LDS stores ----
    for (int e = tid; e < TILE_ROWS * (KCHUNK / 4); e += 256) {
      int k4  = (e & (KCHUNK / 4 - 1)) * 4;   // 0,4,...,60
      int r   = e >> 4;
      int row = rowBase + r;
      int k   = kc + k4;                       // group of 4 never straddles the 128 split
      float4 v = make_float4(0.f, 0.f, 0.f, 0.f);
      if (row < N) {
        if (A1 != nullptr && k >= D128) {
          float inv = 1.f / fmaxf(cntv[row], 1.f);
          float4 t = *(const float4*)(A1 + (size_t)row * D128 + (k - D128));
          v = make_float4(t.x * inv, t.y * inv, t.z * inv, t.w * inv);
        } else {
          float4 t = *(const float4*)(A0 + (size_t)row * D128 + k);
          if (sc0 != nullptr) {
            v.x = fmaxf(0.f, t.x * sc0[k + 0] + sh0[k + 0]);
            v.y = fmaxf(0.f, t.y * sc0[k + 1] + sh0[k + 1]);
            v.z = fmaxf(0.f, t.z * sc0[k + 2] + sh0[k + 2]);
            v.w = fmaxf(0.f, t.w * sc0[k + 3] + sh0[k + 3]);
          } else {
            v = t;
          }
        }
      }
      __bf16* dp = &AL[r * LDSTR + k4];        // 8B-aligned contiguous store
      dp[0] = (__bf16)v.x; dp[1] = (__bf16)v.y;
      dp[2] = (__bf16)v.z; dp[3] = (__bf16)v.w;
    }
    // ---- stage W chunk transposed: coalesced global reads, 8B LDS stores ----
    for (int e = tid; e < D128 * (KCHUNK / 4); e += 256) {
      int c  = e & (D128 - 1);
      int k4 = (e >> 7) * 4;
      __bf16* dp = &WL[c * LDSTR + k4];
      #pragma unroll
      for (int j = 0; j < 4; ++j)
        dp[j] = (__bf16)W[(size_t)(kc + k4 + j) * D128 + c];
    }
    __syncthreads();

    // ---- 2 x K=32 WMMA steps: batch-load all B fragments, then WMMA burst ----
    #pragma unroll
    for (int kb = 0; kb < KCHUNK; kb += 32) {
      // A 16x32 bf16 fragment (ISA 7.12.2 lane-half K interleave)
      const __bf16* ap = &AL[(wave * 16 + l15) * LDSTR + kb + hi * 8];
      v8bf alo = *(const v8bf*)ap;
      v8bf ahi = *(const v8bf*)(ap + 16);
      v16bf a;
      #pragma unroll
      for (int j = 0; j < 8; ++j) { a[j] = alo[j]; a[j + 8] = ahi[j]; }

      v16bf bfr[8];
      #pragma unroll
      for (int t = 0; t < 8; ++t) {
        const __bf16* bp = &WL[(t * 16 + l15) * LDSTR + kb + hi * 8];
        v8bf blo = *(const v8bf*)bp;
        v8bf bhi = *(const v8bf*)(bp + 16);
        #pragma unroll
        for (int j = 0; j < 8; ++j) { bfr[t][j] = blo[j]; bfr[t][j + 8] = bhi[j]; }
      }
      #pragma unroll
      for (int t = 0; t < 8; ++t)
        acc[t] = __builtin_amdgcn_wmma_f32_16x16x32_bf16(
            false, a, false, bfr[t], (short)0, acc[t], false, false);
    }
    __syncthreads();
  }

  // ---- epilogue: bias add, store pre-activation, per-column stats ----
  #pragma unroll
  for (int t = 0; t < 8; ++t) {
    int col  = t * 16 + l15;
    float bv = biasL[col];
    float s = 0.f, ss = 0.f;
    #pragma unroll
    for (int r = 0; r < 8; ++r) {
      int row = rowBase + wave * 16 + hi * 8 + r;   // C layout: VGPR r -> M=r(+8)
      float v = acc[t][r] + bv;
      if (row < N) {
        Yp[(size_t)row * D128 + col] = v;
        s += v; ss += v * v;
      }
    }
    unsafeAtomicAdd(&colS[col], s);
    unsafeAtomicAdd(&colSS[col], ss);
  }
  __syncthreads();
  if (tid < D128) {
    unsafeAtomicAdd(&gsum[tid],   colS[tid]);
    unsafeAtomicAdd(&gsumsq[tid], colSS[tid]);
  }
}

__global__ void finalize_kernel(const float* __restrict__ gsum,
                                const float* __restrict__ gsumsq,
                                const float* __restrict__ g,
                                const float* __restrict__ beta,
                                float* __restrict__ sc, float* __restrict__ sh,
                                float invN)
{
  int c = threadIdx.x;
  float mu  = gsum[c] * invN;
  float var = fmaxf(gsumsq[c] * invN - mu * mu, 0.f);
  float a   = g[c] * rsqrtf(var + 1e-5f);
  sc[c] = a;
  sh[c] = beta[c] - mu * a;
}

// One wave per edge: sums[dst] += BNReLU(H[src]), cnt[dst] += 1
__global__ __launch_bounds__(256)
void scatter_kernel(const float* __restrict__ H,
                    const float* __restrict__ sc, const float* __restrict__ sh,
                    const int* __restrict__ src, const int* __restrict__ dst,
                    float* __restrict__ sums, float* __restrict__ cntv, int E)
{
  long long gid = (long long)blockIdx.x * 256 + threadIdx.x;
  int e    = (int)(gid >> 5);
  int lane = (int)(gid & 31);
  if (e >= E) return;
  int s = src[e], d = dst[e];
  if (lane == 0) unsafeAtomicAdd(&cntv[d], 1.f);
  const float* hp = H + (size_t)s * D128;
  float*       op = sums + (size_t)d * D128;
  #pragma unroll
  for (int i = 0; i < 4; ++i) {
    int c = lane + i * 32;
    float v = fmaxf(0.f, hp[c] * sc[c] + sh[c]);
    unsafeAtomicAdd(&op[c], v);
  }
}

__global__ void bnrelu_out_kernel(const float* __restrict__ P,
                                  const float* __restrict__ sc,
                                  const float* __restrict__ sh,
                                  float* __restrict__ out, long long total)
{
  long long i = (long long)blockIdx.x * 256 + threadIdx.x;
  if (i >= total) return;
  int c = (int)(i & (D128 - 1));
  out[i] = fmaxf(0.f, P[i] * sc[c] + sh[c]);
}

extern "C" void kernel_launch(void* const* d_in, const int* in_sizes, int n_in,
                              void* d_out, int out_size, void* d_ws, size_t ws_size,
                              hipStream_t stream) {
  const float* feat = (const float*)d_in[0];
  const int*   src  = (const int*)d_in[1];
  const int*   dst  = (const int*)d_in[2];
  const float* Wm[6] = { (const float*)d_in[3], (const float*)d_in[4],   // aggr W1,W2
                         (const float*)d_in[5], (const float*)d_in[6],   // self W1,W2
                         (const float*)d_in[7], (const float*)d_in[8] }; // comb W1,W2
  const float* bv[6]; const float* gv[6]; const float* bev[6];
  for (int l = 0; l < 6; ++l) {
    bv[l]  = (const float*)d_in[9 + 3 * l + 0];
    gv[l]  = (const float*)d_in[9 + 3 * l + 1];
    bev[l] = (const float*)d_in[9 + 3 * l + 2];
  }

  const int N = in_sizes[0] / D128;
  const int E = in_sizes[1];
  const long long NB = (long long)N * D128;

  float* X    = (float*)d_ws;            // 51.2 MB
  float* Y    = X + NB;                  // 51.2 MB
  float* cnt  = Y + NB;                  // N floats
  float* st   = cnt + N;                 // stats region: 4 * 6 * 128 floats
  float* sums = (float*)d_out;           // reuse d_out as segment-sum accumulator
  float* outp = (float*)d_out;

  auto SUM = [&](int l) { return st + 0 * 768 + l * 128; };
  auto SQ  = [&](int l) { return st + 1 * 768 + l * 128; };
  auto SC  = [&](int l) { return st + 2 * 768 + l * 128; };
  auto SH  = [&](int l) { return st + 3 * 768 + l * 128; };

  const dim3 blk(256);
  const int  gGemm = (N + TILE_ROWS - 1) / TILE_ROWS;
  const float invN = 1.0f / (float)N;

  // init: zero sums(d_out), cnt, stats
  zero_kernel<<<(unsigned)((NB + 255) / 256), blk, 0, stream>>>(sums, NB);
  zero_kernel<<<(unsigned)((N + 255) / 256), blk, 0, stream>>>(cnt, N);
  zero_kernel<<<(unsigned)((4 * 768 + 255) / 256), blk, 0, stream>>>(st, 4 * 768);

  // ---- aggr MLP ----
  gemm_bn_kernel<<<gGemm, blk, 0, stream>>>(feat, nullptr, nullptr, nullptr, nullptr,
                                            Wm[0], bv[0], X, SUM(0), SQ(0), N, 128);
  finalize_kernel<<<1, 128, 0, stream>>>(SUM(0), SQ(0), gv[0], bev[0], SC(0), SH(0), invN);
  gemm_bn_kernel<<<gGemm, blk, 0, stream>>>(X, SC(0), SH(0), nullptr, nullptr,
                                            Wm[1], bv[1], Y, SUM(1), SQ(1), N, 128);
  finalize_kernel<<<1, 128, 0, stream>>>(SUM(1), SQ(1), gv[1], bev[1], SC(1), SH(1), invN);

  // ---- edge gather + segment sum/count (BN+ReLU of P2 applied inline) ----
  {
    long long thr = (long long)E * 32;
    scatter_kernel<<<(unsigned)((thr + 255) / 256), blk, 0, stream>>>(
        Y, SC(1), SH(1), src, dst, sums, cnt, E);
  }

  // ---- self MLP ----
  gemm_bn_kernel<<<gGemm, blk, 0, stream>>>(feat, nullptr, nullptr, nullptr, nullptr,
                                            Wm[2], bv[2], X, SUM(2), SQ(2), N, 128);
  finalize_kernel<<<1, 128, 0, stream>>>(SUM(2), SQ(2), gv[2], bev[2], SC(2), SH(2), invN);
  gemm_bn_kernel<<<gGemm, blk, 0, stream>>>(X, SC(2), SH(2), nullptr, nullptr,
                                            Wm[3], bv[3], Y, SUM(3), SQ(3), N, 128);
  finalize_kernel<<<1, 128, 0, stream>>>(SUM(3), SQ(3), gv[3], bev[3], SC(3), SH(3), invN);

  // ---- comb layer 1: K=256, concat + segment-mean fused into A loader ----
  gemm_bn_kernel<<<gGemm, blk, 0, stream>>>(Y, SC(3), SH(3), sums, cnt,
                                            Wm[4], bv[4], X, SUM(4), SQ(4), N, 256);
  finalize_kernel<<<1, 128, 0, stream>>>(SUM(4), SQ(4), gv[4], bev[4], SC(4), SH(4), invN);

  // ---- comb layer 2 ----
  gemm_bn_kernel<<<gGemm, blk, 0, stream>>>(X, SC(4), SH(4), nullptr, nullptr,
                                            Wm[5], bv[5], Y, SUM(5), SQ(5), N, 128);
  finalize_kernel<<<1, 128, 0, stream>>>(SUM(5), SQ(5), gv[5], bev[5], SC(5), SH(5), invN);

  // ---- final BN+ReLU -> d_out ----
  bnrelu_out_kernel<<<(unsigned)((NB + 255) / 256), blk, 0, stream>>>(Y, SC(5), SH(5), outp, NB);
}
